// SpatialAttention_56590489092211
// MI455X (gfx1250) — matrix operand order repaired and checked by reading, not compiled, not run on hardware
//
#include <hip/hip_runtime.h>

// ---------------------------------------------------------------------------
// CDNA5 (gfx1250) wave32 WMMA implementation of channel attention with 3x3
// conv projections.  All GEMMs use v_wmma_f32_16x16x32_bf16 (fp32 accum).
// Conv uses a software-pipelined implicit GEMM; A-tiles go through the CDNA5
// async global->LDS path (GLOBAL_LOAD_ASYNC_TO_LDS_B128, ASYNCcnt-tracked).
// ---------------------------------------------------------------------------

typedef __attribute__((ext_vector_type(16))) __bf16 v16bf;
typedef __attribute__((ext_vector_type(8)))  __bf16 v8bf;
typedef __attribute__((ext_vector_type(8)))  float  v8f;
typedef __attribute__((ext_vector_type(4)))  int    v4i;
// 2-byte-aligned vector view for the 3x3-tap-shifted x reads
typedef __attribute__((ext_vector_type(8), aligned(2))) __bf16 v8bf_u;

#if defined(__has_builtin)
#if __has_builtin(__builtin_amdgcn_global_load_async_to_lds_b128) && \
    __has_builtin(__builtin_amdgcn_s_wait_asynccnt)
#define USE_ASYNC_LDS 1
#endif
#endif
#ifndef USE_ASYNC_LDS
#define USE_ASYNC_LDS 0
#endif

#if USE_ASYNC_LDS
typedef __attribute__((address_space(3))) void as3_void;
typedef __attribute__((address_space(1))) void as1_void;
typedef __attribute__((address_space(3))) v4i  as3_v4i;
typedef __attribute__((address_space(1))) v4i  as1_v4i;
#define GLB_V4I(p) ((as1_v4i*)(as1_void*)(p))
#define LDS_V4I(p) ((as3_v4i*)(as3_void*)(p))
#endif

#define CB   512          // channels
#define NB   4096         // H*W
#define KTOT 4608         // C*9 (tap-major K)
#define PAD  66           // 64 + halo

// Build a 16x32 bf16 fragment from two contiguous 16-byte chunks.
// K order used here (contiguous, same for A and B; not AMD's table order):
//   A: lanes 0-15 rows m; elems e -> K = half*16 + e
//   B: column n per lane; elems e -> K = half*16 + e
__device__ __forceinline__ v16bf frag_from_ptrs(const __bf16* lo, const __bf16* hi) {
  v8bf l = *(const v8bf*)lo;
  v8bf h = *(const v8bf*)hi;
  v16bf r;
#pragma unroll
  for (int e = 0; e < 8; ++e) { r[e] = l[e]; r[e + 8] = h[e]; }
  return r;
}

// ---------------------------------------------------------------------------
// Weight repack: w[O][I][3][3] fp32 -> Wp[s][O][tap*512 + i] bf16.
// ---------------------------------------------------------------------------
__global__ __launch_bounds__(256) void pack_weights_kernel(
    const float* __restrict__ wq, const float* __restrict__ wk,
    const float* __restrict__ wv, __bf16* __restrict__ Wp) {
  size_t id = (size_t)blockIdx.x * 256 + threadIdx.x;   // 3*512*4608 total
  int s = (int)(id / ((size_t)CB * KTOT));
  size_t rem = id - (size_t)s * CB * KTOT;
  int o  = (int)(rem / KTOT);
  int kk = (int)(rem % KTOT);
  int r  = kk >> 9;
  int i  = kk & 511;
  const float* src = (s == 0) ? wq : (s == 1) ? wk : wv;
  Wp[id] = (__bf16)src[(size_t)o * KTOT + (size_t)i * 9 + r];
}

// ---------------------------------------------------------------------------
// x fp32 [B][C][64][64] -> bf16 zero-padded [B][C][66][66]
// ---------------------------------------------------------------------------
__global__ __launch_bounds__(256) void pad_x_kernel(
    const float* __restrict__ x, __bf16* __restrict__ xpad) {
  size_t id = (size_t)blockIdx.x * 256 + threadIdx.x;   // 8*512*66*66 total
  int px = (int)(id % PAD);
  size_t rem = id / PAD;
  int py = (int)(rem % PAD);
  size_t bc = rem / PAD;
  __bf16 v = (__bf16)0.0f;
  if (px >= 1 && px <= 64 && py >= 1 && py <= 64)
    v = (__bf16)x[bc * NB + (size_t)(py - 1) * 64 + (px - 1)];
  xpad[id] = v;
}

// ---------------------------------------------------------------------------
// Implicit-GEMM 3x3 conv: Y[m][n] = sum_k Wp[m][k]*im2col(x)[k][n] + bias[m]
// Block tile 128x128, BK=32. 8 waves, each 64x32 (4x2 WMMA accumulators).
// Software pipelined: chunk i+1 loads issued while chunk i computes.
// ---------------------------------------------------------------------------
__global__ __launch_bounds__(256) void conv_wmma_kernel(
    const __bf16* __restrict__ Wp,    // [512][4608] (tap-major K)
    const float*  __restrict__ bias,  // [512]
    const __bf16* __restrict__ xpad,  // [B][512][66][66]
    __bf16* __restrict__ out,         // [B][512][4096] or transposed
    int transpose_out) {
  __shared__ __bf16 As[2][128 * 32];  // [m][k], double buffered for async path
  __shared__ __bf16 Bs[128 * 32];     // [n][k] (transposed for frag loads)

  const int b    = blockIdx.z;
  const int m0   = blockIdx.y * 128;
  const int n0   = blockIdx.x * 128;
  const int t    = threadIdx.x;
  const int wave = t >> 5, lane = t & 31;
  const int half = lane >> 4, idx = lane & 15;
  const int wm   = (wave >> 2) * 64;
  const int wn   = (wave & 3) * 32;
  const __bf16* xb = xpad + (size_t)b * CB * PAD * PAD;

  v8f acc[4][2] = {};

  // A-tile loader geometry: 2 threads per row, 16 elements each
  const int arow = t >> 1;
  const int ahs  = (t & 1) * 16;
  const __bf16* aptr = Wp + (size_t)(m0 + arow) * KTOT + ahs;
  const int alds = arow * 32 + ahs;

  // B-tile task geometry (chunk-independent): 2 tasks x 8 pixels
  int bkk[2], bn8[2], byx[2];
#pragma unroll
  for (int it = 0; it < 2; ++it) {
    int e   = t + it * 256;
    bkk[it] = e >> 4;
    bn8[it] = (e & 15) * 8;
    int gi  = n0 + bn8[it];
    byx[it] = (gi >> 6) * PAD + (gi & 63);   // unshifted (gy*66 + gx)
  }

  v8bf_u bcur[2];
  v8bf   acur[2];

  // ---- preload chunk 0 ----
  {
#pragma unroll
    for (int it = 0; it < 2; ++it)
      bcur[it] = *(const v8bf_u*)(xb + (size_t)bkk[it] * (PAD * PAD) + byx[it]);
#if USE_ASYNC_LDS
    __builtin_amdgcn_global_load_async_to_lds_b128(
        GLB_V4I(aptr), LDS_V4I(&As[0][alds]), 0, 0);
    __builtin_amdgcn_global_load_async_to_lds_b128(
        GLB_V4I(aptr + 8), LDS_V4I(&As[0][alds + 8]), 0, 0);
#else
    acur[0] = *(const v8bf*)(aptr);
    acur[1] = *(const v8bf*)(aptr + 8);
#endif
  }

  for (int k0 = 0, itn = 0; k0 < KTOT; k0 += 32, ++itn) {
    const int buf = itn & 1;
    __syncthreads();
    // ---- commit staged B (and A in sync mode) to LDS ----
#pragma unroll
    for (int j = 0; j < 8; ++j) {
      Bs[(bn8[0] + j) * 32 + bkk[0]] = bcur[0][j];
      Bs[(bn8[1] + j) * 32 + bkk[1]] = bcur[1][j];
    }
#if !USE_ASYNC_LDS
    *(v8bf*)(&As[buf][alds])     = acur[0];
    *(v8bf*)(&As[buf][alds + 8]) = acur[1];
#endif

    // ---- issue next chunk's loads (overlaps with compute below) ----
    const int k1 = k0 + 32;
    if (k1 < KTOT) {
      const int r1  = k1 >> 9;
      const int i1  = k1 & 511;
      const int sh1 = (r1 / 3) * PAD + (r1 % 3);
#pragma unroll
      for (int it = 0; it < 2; ++it)
        bcur[it] = *(const v8bf_u*)(xb + (size_t)(i1 + bkk[it]) * (PAD * PAD)
                                       + byx[it] + sh1);
#if USE_ASYNC_LDS
      __builtin_amdgcn_global_load_async_to_lds_b128(
          GLB_V4I(aptr + k1), LDS_V4I(&As[buf ^ 1][alds]), 0, 0);
      __builtin_amdgcn_global_load_async_to_lds_b128(
          GLB_V4I(aptr + k1 + 8), LDS_V4I(&As[buf ^ 1][alds + 8]), 0, 0);
      __builtin_amdgcn_s_wait_asynccnt(2);   // chunk k0's pair has landed
#endif
      __builtin_prefetch(aptr + k1 + 32, 0, 1);          // A, 2 chunks ahead
      __builtin_prefetch(xb + (size_t)((k1 & 511) + 32 + bkk[0]) * (PAD * PAD)
                            + byx[0], 0, 1);             // B stream hint
    }
#if USE_ASYNC_LDS
    else {
      __builtin_amdgcn_s_wait_asynccnt(0);
    }
#endif
    __syncthreads();

    // ---- fragments + 8 WMMAs ----
    v16bf af[4];
#pragma unroll
    for (int tm = 0; tm < 4; ++tm) {
      const __bf16* rp = &As[buf][(wm + tm * 16 + idx) * 32];
      af[tm] = frag_from_ptrs(rp + half * 16, rp + half * 16 + 8);
    }
    v16bf bfr[2];
#pragma unroll
    for (int tn = 0; tn < 2; ++tn) {
      const __bf16* rp = Bs + (wn + tn * 16 + idx) * 32 + half * 16;
      bfr[tn] = frag_from_ptrs(rp, rp + 8);
    }
#pragma unroll
    for (int tm = 0; tm < 4; ++tm)
#pragma unroll
      for (int tn = 0; tn < 2; ++tn)
        acc[tm][tn] = __builtin_amdgcn_wmma_f32_16x16x32_bf16(
            false, af[tm], false, bfr[tn], (short)0, acc[tm][tn], false, false);
  }

  // ---- epilogue: +bias, store bf16 (transposed for V) ----
#pragma unroll
  for (int tm = 0; tm < 4; ++tm)
#pragma unroll
    for (int tn = 0; tn < 2; ++tn)
#pragma unroll
      for (int rr = 0; rr < 8; ++rr) {
        int m = m0 + wm + tm * 16 + rr + half * 8;
        int n = n0 + wn + tn * 16 + idx;
        float val = acc[tm][tn][rr] + bias[m];
        if (!transpose_out)
          out[(size_t)b * CB * NB + (size_t)m * NB + n] = (__bf16)val;
        else
          out[(size_t)b * NB * CB + (size_t)n * CB + m] = (__bf16)val;
      }
}

// ---------------------------------------------------------------------------
// attn[c][d] = sum_n q[c][n] * k[d][n].  Both row-major [512][4096] bf16:
// B-fragment column d == k-row d, so fragments load straight from global (L2).
// ---------------------------------------------------------------------------
__global__ __launch_bounds__(256) void attn_logits_kernel(
    const __bf16* __restrict__ q, const __bf16* __restrict__ kmat,
    float* __restrict__ attn) {
  const int b    = blockIdx.z;
  const int m0   = blockIdx.y * 128;
  const int n0   = blockIdx.x * 128;
  const int t    = threadIdx.x;
  const int wave = t >> 5, lane = t & 31;
  const int half = lane >> 4, idx = lane & 15;
  const int wm   = (wave >> 2) * 64;
  const int wn   = (wave & 3) * 32;
  const __bf16* qb = q    + (size_t)b * CB * NB;
  const __bf16* kb = kmat + (size_t)b * CB * NB;

  v8f acc[4][2] = {};
#pragma unroll 2
  for (int k0 = 0; k0 < NB; k0 += 32) {
    v16bf af[4], bfr[2];
#pragma unroll
    for (int tm = 0; tm < 4; ++tm) {
      const __bf16* rp = qb + (size_t)(m0 + wm + tm * 16 + idx) * NB + k0;
      af[tm] = frag_from_ptrs(rp + half * 16, rp + half * 16 + 8);
    }
#pragma unroll
    for (int tn = 0; tn < 2; ++tn) {
      const __bf16* rp = kb + (size_t)(n0 + wn + tn * 16 + idx) * NB + k0 + half * 16;
      bfr[tn] = frag_from_ptrs(rp, rp + 8);
    }
#pragma unroll
    for (int tm = 0; tm < 4; ++tm)
#pragma unroll
      for (int tn = 0; tn < 2; ++tn)
        acc[tm][tn] = __builtin_amdgcn_wmma_f32_16x16x32_bf16(
            false, af[tm], false, bfr[tn], (short)0, acc[tm][tn], false, false);
  }
#pragma unroll
  for (int tm = 0; tm < 4; ++tm)
#pragma unroll
    for (int tn = 0; tn < 2; ++tn)
#pragma unroll
      for (int rr = 0; rr < 8; ++rr) {
        int m = m0 + wm + tm * 16 + rr + half * 8;
        int n = n0 + wn + tn * 16 + idx;
        attn[(size_t)b * CB * CB + (size_t)m * CB + n] = acc[tm][tn][rr];
      }
}

// ---------------------------------------------------------------------------
// Row softmax over [B*512] rows of 512, fp32 in -> bf16 out.
// ---------------------------------------------------------------------------
__global__ __launch_bounds__(256) void softmax_kernel(
    const float* __restrict__ attn, __bf16* __restrict__ attn_b) {
  __shared__ float red[256];
  const int row = blockIdx.x;
  const int t   = threadIdx.x;
  const float* src = attn + (size_t)row * CB;
  float v0 = src[t], v1 = src[t + 256];

  red[t] = fmaxf(v0, v1);
  __syncthreads();
  for (int s = 128; s > 0; s >>= 1) {
    if (t < s) red[t] = fmaxf(red[t], red[t + s]);
    __syncthreads();
  }
  float rmax = red[0];
  __syncthreads();

  float e0 = __expf(v0 - rmax), e1 = __expf(v1 - rmax);
  red[t] = e0 + e1;
  __syncthreads();
  for (int s = 128; s > 0; s >>= 1) {
    if (t < s) red[t] += red[t + s];
    __syncthreads();
  }
  float inv = 1.0f / red[0];
  __bf16* dst = attn_b + (size_t)row * CB;
  dst[t]       = (__bf16)(e0 * inv);
  dst[t + 256] = (__bf16)(e1 * inv);
}

// ---------------------------------------------------------------------------
// out[c][n] = gamma * sum_d attn[c][d] * v[d][n] + x[c][n]
// A = attn_b [512][512] row-major, B column n == vT row n (contiguous).
// ---------------------------------------------------------------------------
__global__ __launch_bounds__(256) void attn_apply_kernel(
    const __bf16* __restrict__ attn_b,  // [B][512][512]
    const __bf16* __restrict__ vT,      // [B][4096][512]
    const float*  __restrict__ x,       // [B][512][4096]
    const float*  __restrict__ gamma,   // [1]
    float* __restrict__ out) {
  const int b    = blockIdx.z;
  const int m0   = blockIdx.y * 128;
  const int n0   = blockIdx.x * 128;
  const int t    = threadIdx.x;
  const int wave = t >> 5, lane = t & 31;
  const int half = lane >> 4, idx = lane & 15;
  const int wm   = (wave >> 2) * 64;
  const int wn   = (wave & 3) * 32;
  const __bf16* ab = attn_b + (size_t)b * CB * CB;
  const __bf16* vb = vT     + (size_t)b * NB * CB;

  v8f acc[4][2] = {};
#pragma unroll 2
  for (int k0 = 0; k0 < CB; k0 += 32) {
    v16bf af[4], bfr[2];
#pragma unroll
    for (int tm = 0; tm < 4; ++tm) {
      const __bf16* rp = ab + (size_t)(m0 + wm + tm * 16 + idx) * CB + k0;
      af[tm] = frag_from_ptrs(rp + half * 16, rp + half * 16 + 8);
    }
#pragma unroll
    for (int tn = 0; tn < 2; ++tn) {
      const __bf16* rp = vb + (size_t)(n0 + wn + tn * 16 + idx) * CB + k0 + half * 16;
      bfr[tn] = frag_from_ptrs(rp, rp + 8);
    }
#pragma unroll
    for (int tm = 0; tm < 4; ++tm)
#pragma unroll
      for (int tn = 0; tn < 2; ++tn)
        acc[tm][tn] = __builtin_amdgcn_wmma_f32_16x16x32_bf16(
            false, af[tm], false, bfr[tn], (short)0, acc[tm][tn], false, false);
  }
  const float g = gamma[0];
#pragma unroll
  for (int tm = 0; tm < 4; ++tm)
#pragma unroll
    for (int tn = 0; tn < 2; ++tn)
#pragma unroll
      for (int rr = 0; rr < 8; ++rr) {
        int m = m0 + wm + tm * 16 + rr + half * 8;
        int n = n0 + wn + tn * 16 + idx;
        size_t o = (size_t)b * CB * NB + (size_t)m * NB + n;
        out[o] = g * acc[tm][tn][rr] + x[o];
      }
}

// ---------------------------------------------------------------------------
extern "C" void kernel_launch(void* const* d_in, const int* in_sizes, int n_in,
                              void* d_out, int out_size, void* d_ws, size_t ws_size,
                              hipStream_t stream) {
  const float* x     = (const float*)d_in[0];
  const float* wq    = (const float*)d_in[1];
  const float* bq    = (const float*)d_in[2];
  const float* wk    = (const float*)d_in[3];
  const float* bk    = (const float*)d_in[4];
  const float* wv    = (const float*)d_in[5];
  const float* bv    = (const float*)d_in[6];
  const float* gamma = (const float*)d_in[7];
  float* out = (float*)d_out;

  // workspace carve-up (~163 MB)
  char* ws = (char*)d_ws;
  __bf16* Wp    = (__bf16*)ws;  ws += 3ull * CB * KTOT * 2;        // 14.2 MB
  __bf16* xpad  = (__bf16*)ws;  ws += 8ull * CB * PAD * PAD * 2;   // 35.7 MB
  __bf16* qb    = (__bf16*)ws;  ws += 8ull * CB * NB * 2;          // 33.6 MB
  __bf16* kbuf  = (__bf16*)ws;  ws += 8ull * CB * NB * 2;          // 33.6 MB
  __bf16* vT    = (__bf16*)ws;  ws += 8ull * NB * CB * 2;          // 33.6 MB
  float*  attn  = (float*)ws;   ws += 8ull * CB * CB * 4;          // 8.4 MB
  __bf16* attnb = (__bf16*)ws;                                     // 4.2 MB

  pack_weights_kernel<<<27648, 256, 0, stream>>>(wq, wk, wv, Wp);
  pad_x_kernel<<<69696, 256, 0, stream>>>(x, xpad);

  dim3 cg(32, 4, 8);   // N-tiles, M-tiles, batch
  conv_wmma_kernel<<<cg, 256, 0, stream>>>(Wp,                    bq, xpad, qb,   0);
  conv_wmma_kernel<<<cg, 256, 0, stream>>>(Wp + 1ull * CB * KTOT, bk, xpad, kbuf, 0);
  conv_wmma_kernel<<<cg, 256, 0, stream>>>(Wp + 2ull * CB * KTOT, bv, xpad, vT,   1);

  attn_logits_kernel<<<dim3(4, 4, 8), 256, 0, stream>>>(qb, kbuf, attn);
  softmax_kernel<<<4096, 256, 0, stream>>>(attn, attnb);
  attn_apply_kernel<<<dim3(32, 4, 8), 256, 0, stream>>>(attnb, vT, x, gamma, out);
}
